// TemporalSelfAttention_90202903150982
// MI455X (gfx1250) — compile-verified, matrix operand order checked by reading, not run on hardware
//
#include <hip/hip_runtime.h>

typedef __bf16 bf16;
typedef __attribute__((ext_vector_type(8)))  bf16  bf16x8;
typedef __attribute__((ext_vector_type(16))) bf16  bf16x16;
typedef __attribute__((ext_vector_type(8)))  float floatx8;
typedef __attribute__((ext_vector_type(4)))  float floatx4;

static constexpr int B_ = 4, T_ = 1024, E_ = 4096, HEADS = 8, DH = 512;
static constexpr float SCALE = 0.015625f; // 4096^-0.5

#define USE_ASYNC_LDS 1

static __device__ __forceinline__ floatx8 wmma_bf16(bf16x16 a, bf16x16 b, floatx8 c) {
    // v_wmma_f32_16x16x32_bf16: (neg_a, A, neg_b, B, c_mod, C, reuse_a, reuse_b)
    return __builtin_amdgcn_wmma_f32_16x16x32_bf16(false, a, false, b, (short)0, c, false, false);
}

static __device__ __forceinline__ bf16x16 cat16(bf16x8 lo, bf16x8 hi) {
    bf16x16 v;
#pragma unroll
    for (int i = 0; i < 8; ++i) { v[i] = lo[i]; v[i + 8] = hi[i]; }
    return v;
}

#if USE_ASYNC_LDS
// CDNA5 async DMA: copy 16B per lane from global to LDS, tracked by ASYNCcnt.
static __device__ __forceinline__ void async_b128_to_lds(const void* gaddr, void* lds) {
    unsigned           l = (unsigned)(size_t)lds;               // LDS byte offset
    unsigned long long g = (unsigned long long)(size_t)gaddr;   // global address
    asm volatile("global_load_async_to_lds_b128 %0, %1, off"
                 :: "v"(l), "v"(g) : "memory");
}
static __device__ __forceinline__ void wait_asynccnt0() {
    asm volatile("s_wait_asynccnt 0x0" ::: "memory");
}
#endif

// ---------------------------------------------------------------------------
// Elementwise fp32 -> bf16 conversion (one-time; makes weights L2-resident).
// ---------------------------------------------------------------------------
__global__ __launch_bounds__(256)
void cvt_f32_bf16_kernel(const float* __restrict__ src, bf16* __restrict__ dst) {
    size_t i = ((size_t)blockIdx.x * 256 + threadIdx.x) * 8;
    const floatx4* p = (const floatx4*)(src + i);
    floatx4 a = p[0], b = p[1];
    bf16x8 r;
#pragma unroll
    for (int j = 0; j < 4; ++j) { r[j] = (bf16)a[j]; r[4 + j] = (bf16)b[j]; }
    *(bf16x8*)(dst + i) = r;
}

// ---------------------------------------------------------------------------
// Tiled WMMA GEMM: C[M,N] = A[M,K] @ B[K,N] (+ bias), bf16 in / f32 accum.
// BM=128, BN=128, BK=32; 256 threads = 8 waves, wave tile 32x64 (2x4 frags).
// A tile staged via async global->LDS DMA; B tile transposed through VGPRs.
// ---------------------------------------------------------------------------
template <typename TC, bool BIAS>
__global__ __launch_bounds__(256)
void gemm_wmma_kernel(const bf16* __restrict__ A, const bf16* __restrict__ B,
                      const float* __restrict__ bias, TC* __restrict__ C,
                      int M, int N, int K) {
    constexpr int BM = 128, BN = 128, BK = 32, LDT = BK + 8;
    __shared__ bf16 As[BM][LDT];
    __shared__ bf16 Bs[BN][LDT]; // transposed: [n][k]

    const int tid  = threadIdx.x;
    const int lane = tid & 31;
    const int wid  = tid >> 5;
    const int wm   = wid & 3;  // 4 row-strips of 32
    const int wn   = wid >> 2; // 2 col-strips of 64
    const int r    = lane & 15;
    const int kh   = (lane >> 4) * 8;
    const int rowhalf = (lane >> 4) * 8;

    const int bm = blockIdx.y * BM;
    const int bn = blockIdx.x * BN;

    floatx8 acc[2][4];
#pragma unroll
    for (int i = 0; i < 2; ++i)
#pragma unroll
        for (int j = 0; j < 4; ++j) acc[i][j] = (floatx8){};

    for (int k0 = 0; k0 < K; k0 += BK) {
        // Stage A tile (128 x 32 bf16, padded rows) -- async DMA, no VGPR hop
#pragma unroll
        for (int it = 0; it < 2; ++it) {
            int c   = tid + it * 256;      // 512 chunks of 8 bf16 (16B)
            int row = c >> 2;
            int kc  = (c & 3) * 8;
#if USE_ASYNC_LDS
            async_b128_to_lds(A + (size_t)(bm + row) * K + k0 + kc, &As[row][kc]);
#else
            *(bf16x8*)&As[row][kc] =
                *(const bf16x8*)(A + (size_t)(bm + row) * K + k0 + kc);
#endif
        }
        // Stage B tile (32 x 128) transposed into Bs[n][k]
#pragma unroll
        for (int it = 0; it < 2; ++it) {
            int c  = tid + it * 256;       // 512 chunks of 8 bf16
            int kk = c >> 4;
            int nc = (c & 15) * 8;
            bf16x8 v = *(const bf16x8*)(B + (size_t)(k0 + kk) * N + bn + nc);
#pragma unroll
            for (int j = 0; j < 8; ++j) Bs[nc + j][kk] = v[j];
        }
#if USE_ASYNC_LDS
        wait_asynccnt0();
#endif
        __syncthreads();

        bf16x16 af[2], bfr[4];
#pragma unroll
        for (int i = 0; i < 2; ++i) {
            int row = wm * 32 + i * 16 + r;
            af[i] = cat16(*(const bf16x8*)&As[row][kh],
                          *(const bf16x8*)&As[row][kh + 16]);
        }
#pragma unroll
        for (int j = 0; j < 4; ++j) {
            int col = wn * 64 + j * 16 + r;
            bfr[j] = cat16(*(const bf16x8*)&Bs[col][kh],
                           *(const bf16x8*)&Bs[col][kh + 16]);
        }
#pragma unroll
        for (int i = 0; i < 2; ++i)
#pragma unroll
            for (int j = 0; j < 4; ++j)
                acc[i][j] = wmma_bf16(af[i], bfr[j], acc[i][j]);
        __syncthreads();
    }

    // Epilogue: C-frag element (vj, lane) -> (M = vj + 8*(lane>>4), N = lane&15)
#pragma unroll
    for (int i = 0; i < 2; ++i) {
#pragma unroll
        for (int j = 0; j < 4; ++j) {
            int n = bn + wn * 64 + j * 16 + r;
            float bv = 0.0f;
            if constexpr (BIAS) bv = bias[n];
#pragma unroll
            for (int vj = 0; vj < 8; ++vj) {
                int m = bm + wm * 32 + i * 16 + rowhalf + vj;
                C[(size_t)m * N + n] = (TC)(acc[i][j][vj] + bv);
            }
        }
    }
}

// ---------------------------------------------------------------------------
// Flash-style attention: one (b,h) slice per blockIdx group; 4 waves x 16 rows.
// qkv: bf16 [B*T][3E] (q | k | v, head h at column h*DH). Output bf16 [B*T][E].
// ---------------------------------------------------------------------------
__global__ __launch_bounds__(128)
void attn_kernel(const bf16* __restrict__ qkv, bf16* __restrict__ attn_out) {
    __shared__ bf16 v_s[DH][40];       // V block transposed: [d][s], 32 keys
    __shared__ bf16 p_s[4][16][40];    // per-wave P staging (16x32)

    const int tid     = threadIdx.x;
    const int lane    = tid & 31;
    const int w       = tid >> 5;
    const int r       = lane & 15;
    const int kh      = (lane >> 4) * 8;
    const int rowhalf = (lane >> 4) * 8;

    const int blk    = blockIdx.x;
    const int b      = blk >> 7;        // / (HEADS * T/64)
    const int h      = (blk >> 4) & 7;
    const int rowblk = blk & 15;
    const int t0     = rowblk * 64 + w * 16;

    const size_t rs = 3 * (size_t)E_;
    const bf16* qbase = qkv + (size_t)b * T_ * rs + (size_t)h * DH;
    const bf16* kbase = qbase + E_;
    const bf16* vbase = qbase + 2 * E_;

    // Preload Q fragments for this wave's 16 rows: 16 chunks of K=32 over dh=512
    bf16x16 qf[16];
#pragma unroll
    for (int d = 0; d < 16; ++d) {
        const bf16* p = qbase + (size_t)(t0 + r) * rs + d * 32 + kh;
        qf[d] = cat16(*(const bf16x8*)p, *(const bf16x8*)(p + 16));
    }

    floatx8 of[32];
#pragma unroll
    for (int j = 0; j < 32; ++j) of[j] = (floatx8){};
    float mrun[8], lrun[8];
#pragma unroll
    for (int vj = 0; vj < 8; ++vj) { mrun[vj] = -1e30f; lrun[vj] = 0.0f; }

    for (int s0 = 0; s0 < T_; s0 += 32) {
        __syncthreads(); // protect v_s / p_s reuse across iterations
        // Cooperative V load, transposed into v_s[d][s]
#pragma unroll
        for (int it = 0; it < 16; ++it) {
            int c  = tid + it * 128;   // 2048 chunks of 8 (contig in d)
            int s  = c >> 6;
            int dc = (c & 63) * 8;
            bf16x8 v = *(const bf16x8*)(vbase + (size_t)(s0 + s) * rs + dc);
#pragma unroll
            for (int j = 0; j < 8; ++j) v_s[dc + j][s] = v[j];
        }
        __syncthreads();

        // S = Q K^T for 16 queries x 32 keys (two 16-col fragments)
        floatx8 sf[2];
        sf[0] = (floatx8){}; sf[1] = (floatx8){};
#pragma unroll
        for (int d = 0; d < 16; ++d) {
#pragma unroll
            for (int jn = 0; jn < 2; ++jn) {
                const bf16* p = kbase + (size_t)(s0 + jn * 16 + r) * rs + d * 32 + kh;
                bf16x16 kf = cat16(*(const bf16x8*)p, *(const bf16x8*)(p + 16));
                sf[jn] = wmma_bf16(qf[d], kf, sf[jn]);
            }
        }

        // Online softmax (fp32) + stage P as bf16 in LDS
#pragma unroll
        for (int vj = 0; vj < 8; ++vj) {
            float x0 = sf[0][vj] * SCALE;
            float x1 = sf[1][vj] * SCALE;
            float mx = fmaxf(x0, x1);
#pragma unroll
            for (int off = 8; off > 0; off >>= 1)
                mx = fmaxf(mx, __shfl_xor(mx, off, 16));
            float mnew  = fmaxf(mrun[vj], mx);
            float alpha = __expf(mrun[vj] - mnew);
            float p0 = __expf(x0 - mnew);
            float p1 = __expf(x1 - mnew);
            float rsum = p0 + p1;
#pragma unroll
            for (int off = 8; off > 0; off >>= 1)
                rsum += __shfl_xor(rsum, off, 16);
            lrun[vj] = lrun[vj] * alpha + rsum;
            mrun[vj] = mnew;
#pragma unroll
            for (int j = 0; j < 32; ++j) of[j][vj] *= alpha;
            p_s[w][rowhalf + vj][r]      = (bf16)p0;
            p_s[w][rowhalf + vj][16 + r] = (bf16)p1;
        }
        __syncthreads();

        // O += P @ V : A = P (16x32), B = V chunk (32 x 16), 32 dh-chunks
        const bf16* pp = &p_s[w][r][kh];
        bf16x16 pf = cat16(*(const bf16x8*)pp, *(const bf16x8*)(pp + 16));
#pragma unroll
        for (int j = 0; j < 32; ++j) {
            const bf16* vp = &v_s[j * 16 + r][kh];
            bf16x16 vf = cat16(*(const bf16x8*)vp, *(const bf16x8*)(vp + 16));
            of[j] = wmma_bf16(pf, vf, of[j]);
        }
    }

    // Normalize and store (bf16) into attn_out[b*T + t][h*DH + d]
#pragma unroll
    for (int vj = 0; vj < 8; ++vj) {
        float inv = 1.0f / lrun[vj];
#pragma unroll
        for (int j = 0; j < 32; ++j) of[j][vj] *= inv;
    }
    bf16* obase = attn_out + ((size_t)b * T_ + t0) * E_ + (size_t)h * DH;
#pragma unroll
    for (int j = 0; j < 32; ++j) {
#pragma unroll
        for (int vj = 0; vj < 8; ++vj) {
            obase[(size_t)(rowhalf + vj) * E_ + j * 16 + r] = (bf16)of[j][vj];
        }
    }
}

extern "C" void kernel_launch(void* const* d_in, const int* in_sizes, int n_in,
                              void* d_out, int out_size, void* d_ws, size_t ws_size,
                              hipStream_t stream) {
    (void)in_sizes; (void)n_in; (void)out_size; (void)ws_size;
    const float* x    = (const float*)d_in[0]; // [B,T,C,H,W] == [4096][4096]
    const float* Wqkv = (const float*)d_in[1]; // [4096][12288]
    const float* Wout = (const float*)d_in[2]; // [4096][4096]
    const float* bout = (const float*)d_in[3]; // [4096]
    float* out = (float*)d_out;                // [4096][4096] fp32

    const size_t n_x    = (size_t)(B_ * T_) * E_;       // 16.8M
    const size_t n_wqkv = (size_t)E_ * 3 * E_;          // 50.3M
    const size_t n_wout = (size_t)E_ * E_;              // 16.8M
    const size_t n_qkv  = (size_t)(B_ * T_) * 3 * E_;   // 50.3M

    char* wsb = (char*)d_ws;
    bf16* xb    = (bf16*)wsb;                 wsb += n_x    * sizeof(bf16); // 32 MB
    bf16* wqkvb = (bf16*)wsb;                 wsb += n_wqkv * sizeof(bf16); // 96 MB (fits L2)
    bf16* woutb = (bf16*)wsb;                 wsb += n_wout * sizeof(bf16); // 32 MB
    bf16* qkv   = (bf16*)wsb;                 wsb += n_qkv  * sizeof(bf16); // 96 MB
    bf16* attn  = (bf16*)wsb;                                               // 32 MB

    // 0) one-time fp32 -> bf16 conversion (weights become L2-resident)
    cvt_f32_bf16_kernel<<<dim3((unsigned)(n_x    / 2048)), 256, 0, stream>>>(x,    xb);
    cvt_f32_bf16_kernel<<<dim3((unsigned)(n_wqkv / 2048)), 256, 0, stream>>>(Wqkv, wqkvb);
    cvt_f32_bf16_kernel<<<dim3((unsigned)(n_wout / 2048)), 256, 0, stream>>>(Wout, woutb);

    // 1) qkv = x @ Wqkv  (M=4096, N=12288, K=4096), bf16 output
    gemm_wmma_kernel<bf16, false>
        <<<dim3((3 * E_) / 128, (B_ * T_) / 128), 256, 0, stream>>>(
            xb, wqkvb, nullptr, qkv, B_ * T_, 3 * E_, E_);

    // 2) flash attention per (b, h, 64-row block)
    attn_kernel<<<dim3(B_ * HEADS * (T_ / 64)), 128, 0, stream>>>(qkv, attn);

    // 3) out = attn @ Wout + bout  (M=N=K=4096), fp32 output
    gemm_wmma_kernel<float, true>
        <<<dim3(E_ / 128, (B_ * T_) / 128), 256, 0, stream>>>(
            attn, woutb, bout, out, B_ * T_, E_, E_);
}
